// AttentionHead_16870631538698
// MI455X (gfx1250) — matrix operand order repaired and checked by reading, not compiled, not run on hardware
//
#include <hip/hip_runtime.h>
#include <hip/hip_fp16.h>

typedef __attribute__((ext_vector_type(16))) _Float16 v16h;
typedef __attribute__((ext_vector_type(8)))  _Float16 v8h;
typedef __attribute__((ext_vector_type(8)))  float    v8f;

#define B_SZ  8
#define N_SEQ 2048
#define D_DIM 768
#define H_DIM 64

static __device__ __forceinline__ v8f wmma_f16(v16h a, v16h b, v8f c) {
  // D = A(16x32 f16) * B(32x16 f16) + C(16x16 f32)
  return __builtin_amdgcn_wmma_f32_16x16x32_f16(false, a, false, b, (short)0, c,
                                                false, false);
}

// ---------------------------------------------------------------------------
// Kernel 0: one-time weight prep.  Wq/Wk/Wv [768][64] f32 -> Wt[3][64][768] f16
// (transposed so WMMA B-fragments are contiguous 16-byte loads).
// ---------------------------------------------------------------------------
__global__ __launch_bounds__(256, 1)
void wcvt_kernel(const float* __restrict__ Wq,
                 const float* __restrict__ Wk,
                 const float* __restrict__ Wv,
                 _Float16* __restrict__ Wt)
{
  const int idx = blockIdx.x * 256 + threadIdx.x;   // over 3*64*768
  if (idx >= 3 * H_DIM * D_DIM) return;
  const int m   = idx / (H_DIM * D_DIM);
  const int rem = idx % (H_DIM * D_DIM);
  const int h   = rem / D_DIM;
  const int d   = rem % D_DIM;
  const float* W = (m == 0) ? Wq : (m == 1) ? Wk : Wv;
  Wt[idx] = (_Float16)W[(size_t)d * H_DIM + h];
}

// ---------------------------------------------------------------------------
// Kernel A: QKV projection.  X[B*N,768] (f32, staged once per block into LDS
// as f16) @ Wt (f16, transposed) -> Q,K row-major f16 and V transposed f16.
// Grid: (B*N)/16 blocks of 128 threads (4 waves); wave ct owns H-cols 16ct..+15.
// ---------------------------------------------------------------------------
__global__ __launch_bounds__(128, 1)
void qkv_proj_kernel(const float* __restrict__ X,
                     const _Float16* __restrict__ Wt,
                     _Float16* __restrict__ Qh,
                     _Float16* __restrict__ Kh,
                     _Float16* __restrict__ Vt)
{
  // 16 rows x 768 halves, padded to 776 (stride 1552B -> rows hit distinct
  // bank groups, 16B-aligned chunks).
  __shared__ __align__(16) _Float16 Xs[16][776];

  const int lane = threadIdx.x & 31;
  const int ct   = threadIdx.x >> 5;      // 0..3 : 16-wide column tile of H
  const int l15  = lane & 15;
  const int hi   = lane >> 4;             // half-wave select
  const int rowbase = blockIdx.x << 4;    // flattened over B*N
  const int abase = hi * 8;               // A-frag K base (16-bit layout)
  const int bbase = hi * 8;               // B-frag K base
  const int col   = ct * 16 + l15;

  // ---- cooperative stage: X tile f32 -> f16 in LDS (once per block) ----
  {
    const float* xt = X + (size_t)rowbase * D_DIM;
    for (int i = threadIdx.x; i < 16 * (D_DIM / 4); i += 128) {
      const int row = i / (D_DIM / 4);
      const int c4  = (i % (D_DIM / 4)) * 4;
      float4 f = *(const float4*)(xt + (size_t)row * D_DIM + c4);
      _Float16* p = &Xs[row][c4];
      p[0] = (_Float16)f.x; p[1] = (_Float16)f.y;
      p[2] = (_Float16)f.z; p[3] = (_Float16)f.w;
    }
  }
  __syncthreads();

  v8f acc[3];
  #pragma unroll
  for (int m = 0; m < 3; ++m)
    #pragma unroll
    for (int r = 0; r < 8; ++r) acc[m][r] = 0.0f;

  const _Float16* wcol[3] = {
    Wt + 0 * (H_DIM * D_DIM) + (size_t)col * D_DIM + bbase,
    Wt + 1 * (H_DIM * D_DIM) + (size_t)col * D_DIM + bbase,
    Wt + 2 * (H_DIM * D_DIM) + (size_t)col * D_DIM + bbase
  };

  for (int kk = 0; kk < D_DIM; kk += 32) {
    // A fragment from LDS: row (lane&15), K = abase+{0..7} and abase+16+{0..7}
    v8h a0 = *(const v8h*)(&Xs[l15][kk + abase]);
    v8h a1 = *(const v8h*)(&Xs[l15][kk + abase + 16]);
    v16h a;
    #pragma unroll
    for (int j = 0; j < 8; ++j) { a[j] = a0[j]; a[8 + j] = a1[j]; }

    // B fragments: lane holds W column `col`, K = bbase+{0..7} and bbase+16+{0..7}
    #pragma unroll
    for (int m = 0; m < 3; ++m) {
      const _Float16* w = wcol[m] + kk;
      v8h b0 = *(const v8h*)(w);
      v8h b1 = *(const v8h*)(w + 16);
      v16h bf;
      #pragma unroll
      for (int j = 0; j < 8; ++j) { bf[j] = b0[j]; bf[8 + j] = b1[j]; }
      acc[m] = wmma_f16(a, bf, acc[m]);
    }
  }

  const int bidx = rowbase / N_SEQ;
  const int n0   = rowbase % N_SEQ;
  #pragma unroll
  for (int r = 0; r < 8; ++r) {
    const int M = r + abase;                       // C layout: M = r + 8*hi
    const size_t g = (size_t)(rowbase + M) * H_DIM + col;
    Qh[g] = (_Float16)acc[0][r];
    Kh[g] = (_Float16)acc[1][r];
    Vt[(size_t)bidx * H_DIM * N_SEQ + (size_t)col * N_SEQ + (size_t)(n0 + M)] =
        (_Float16)acc[2][r];
  }
}

// ---------------------------------------------------------------------------
// Kernel B: causal flash attention.  One wave per 16-query tile, 8 waves/block.
// Grid: (N/128, B).  Online softmax over 32-key blocks; 8 WMMA per key block.
// ---------------------------------------------------------------------------
__global__ __launch_bounds__(256, 1)
void flash_attn_kernel(const _Float16* __restrict__ Qh,
                       const _Float16* __restrict__ Kh,
                       const _Float16* __restrict__ Vt,
                       float* __restrict__ Out)
{
  // Per-wave 16x32 f16 P tile, padded to 40 halves/row -> bank-conflict-free.
  __shared__ __align__(16) _Float16 Plds[8][16][40];

  const int lane = threadIdx.x & 31;
  const int wave = threadIdx.x >> 5;        // 0..7 : query tile within block
  const int l15  = lane & 15;
  const int hi   = lane >> 4;
  const int abase = hi * 8;
  const int bbase = hi * 8;

  const int b     = blockIdx.y;
  const int qbase = blockIdx.x * 128 + wave * 16;

  const _Float16* Qb = Qh + (size_t)b * N_SEQ * H_DIM;
  const _Float16* Kb = Kh + (size_t)b * N_SEQ * H_DIM;
  const _Float16* Vb = Vt + (size_t)b * H_DIM * N_SEQ;

  // Q A-fragments for the two 32-wide chunks of H
  v16h qa[2];
  {
    const _Float16* qr = Qb + (size_t)(qbase + l15) * H_DIM;
    #pragma unroll
    for (int c = 0; c < 2; ++c) {
      v8h p0 = *(const v8h*)(qr + 32 * c + abase);
      v8h p1 = *(const v8h*)(qr + 32 * c + abase + 16);
      #pragma unroll
      for (int j = 0; j < 8; ++j) { qa[c][j] = p0[j]; qa[c][8 + j] = p1[j]; }
    }
  }

  v8f o[4];
  float m[8], l[8];
  #pragma unroll
  for (int c4 = 0; c4 < 4; ++c4)
    #pragma unroll
    for (int r = 0; r < 8; ++r) o[c4][r] = 0.0f;
  #pragma unroll
  for (int r = 0; r < 8; ++r) { m[r] = -__builtin_inff(); l[r] = 0.0f; }

  const int nkb = (qbase + 16 + 31) >> 5;   // key blocks covering causal range
  for (int kb = 0; kb < nkb; ++kb) {
    const int kbase = kb * 32;
    if (kb + 1 < nkb)
      __builtin_prefetch(Kb + (size_t)(kbase + 32 + lane) * H_DIM, 0, 1);

    // ---- S = Q K^T : 16 queries x 32 keys (two 16x16 C tiles) ----
    v8f s[2];
    #pragma unroll
    for (int t = 0; t < 2; ++t) {
      const _Float16* kr = Kb + (size_t)(kbase + 16 * t + l15) * H_DIM;
      v8f accs;
      #pragma unroll
      for (int r = 0; r < 8; ++r) accs[r] = 0.0f;
      #pragma unroll
      for (int c = 0; c < 2; ++c) {
        v8h p0 = *(const v8h*)(kr + 32 * c + bbase);
        v8h p1 = *(const v8h*)(kr + 32 * c + bbase + 16);
        v16h bf;
        #pragma unroll
        for (int j = 0; j < 8; ++j) { bf[j] = p0[j]; bf[8 + j] = p1[j]; }
        accs = wmma_f16(qa[c], bf, accs);
      }
      s[t] = accs;
    }

    // ---- causal mask BEFORE scaling (matches reference) ----
    if (kbase + 31 > qbase) {
      #pragma unroll
      for (int t = 0; t < 2; ++t) {
        const int key = kbase + 16 * t + l15;
        #pragma unroll
        for (int r = 0; r < 8; ++r) {
          const int q = qbase + r + abase;
          if (key > q) s[t][r] = -__builtin_inff();
        }
      }
    }
    #pragma unroll
    for (int t = 0; t < 2; ++t)
      #pragma unroll
      for (int r = 0; r < 8; ++r) s[t][r] *= 0.125f;   // 1/sqrt(64)

    // ---- online softmax; each C row lives in one VGPR across 16 lanes ----
    float corr[8];
    #pragma unroll
    for (int r = 0; r < 8; ++r) {
      float mx = fmaxf(s[0][r], s[1][r]);
      #pragma unroll
      for (int off = 1; off < 16; off <<= 1)
        mx = fmaxf(mx, __shfl_xor(mx, off, 16));
      const float mnew = fmaxf(m[r], mx);
      const float c0 = __expf(m[r] - mnew);
      const float p0 = __expf(s[0][r] - mnew);
      const float p1 = __expf(s[1][r] - mnew);
      s[0][r] = p0; s[1][r] = p1;
      float rs = p0 + p1;
      #pragma unroll
      for (int off = 1; off < 16; off <<= 1)
        rs += __shfl_xor(rs, off, 16);
      l[r] = l[r] * c0 + rs;
      m[r] = mnew;
      corr[r] = c0;
    }
    #pragma unroll
    for (int c4 = 0; c4 < 4; ++c4)
      #pragma unroll
      for (int r = 0; r < 8; ++r) o[c4][r] *= corr[r];

    // ---- P (f32, C layout) -> LDS (f16) -> A-fragment layout ----
    #pragma unroll
    for (int t = 0; t < 2; ++t)
      #pragma unroll
      for (int r = 0; r < 8; ++r)
        Plds[wave][r + abase][16 * t + l15] = (_Float16)s[t][r];
    asm volatile("s_wait_dscnt 0x0" ::: "memory");
    v16h pa;
    {
      const _Float16* pr = &Plds[wave][l15][0];
      v8h c0 = *(const v8h*)(pr + abase);
      v8h c1 = *(const v8h*)(pr + abase + 16);
      #pragma unroll
      for (int j = 0; j < 8; ++j) { pa[j] = c0[j]; pa[8 + j] = c1[j]; }
    }

    // ---- O += P V  (V^T rows give contiguous B fragments) ----
    #pragma unroll
    for (int c4 = 0; c4 < 4; ++c4) {
      const _Float16* vr = Vb + (size_t)(16 * c4 + l15) * N_SEQ + kbase + bbase;
      v8h p0 = *(const v8h*)(vr);
      v8h p1 = *(const v8h*)(vr + 16);
      v16h bf;
      #pragma unroll
      for (int j = 0; j < 8; ++j) { bf[j] = p0[j]; bf[8 + j] = p1[j]; }
      o[c4] = wmma_f16(pa, bf, o[c4]);
    }
  }

  // ---- finalize: Out = O / l  (fp32 output) ----
  float invl[8];
  #pragma unroll
  for (int r = 0; r < 8; ++r) invl[r] = 1.0f / l[r];
  #pragma unroll
  for (int c4 = 0; c4 < 4; ++c4)
    #pragma unroll
    for (int r = 0; r < 8; ++r) {
      const size_t q = (size_t)(qbase + r + abase);
      Out[((size_t)b * N_SEQ + q) * H_DIM + 16 * c4 + l15] = o[c4][r] * invl[r];
    }
}

// ---------------------------------------------------------------------------
extern "C" void kernel_launch(void* const* d_in, const int* in_sizes, int n_in,
                              void* d_out, int out_size, void* d_ws, size_t ws_size,
                              hipStream_t stream) {
  (void)in_sizes; (void)n_in; (void)out_size; (void)ws_size;
  const float* X  = (const float*)d_in[0];
  const float* Wq = (const float*)d_in[1];
  const float* Wk = (const float*)d_in[2];
  const float* Wv = (const float*)d_in[3];
  float* Out = (float*)d_out;

  const size_t nQ = (size_t)B_SZ * N_SEQ * H_DIM;   // 1M elems each
  _Float16* Qh = (_Float16*)d_ws;                   // [B*N][64]
  _Float16* Kh = Qh + nQ;                           // [B*N][64]
  _Float16* Vt = Kh + nQ;                           // [B][64][N]
  _Float16* Wt = Vt + nQ;                           // [3][64][768]

  wcvt_kernel<<<dim3((3 * H_DIM * D_DIM + 255) / 256), dim3(256), 0, stream>>>(
      Wq, Wk, Wv, Wt);
  qkv_proj_kernel<<<dim3((B_SZ * N_SEQ) / 16), dim3(128), 0, stream>>>(
      X, Wt, Qh, Kh, Vt);
  flash_attn_kernel<<<dim3(N_SEQ / 128, B_SZ), dim3(256), 0, stream>>>(
      Qh, Kh, Vt, Out);
}
